// MyNet_76622216560934
// MI455X (gfx1250) — matrix operand order repaired and checked by reading, not compiled, not run on hardware
//
#include <hip/hip_runtime.h>
#include <hip/hip_bf16.h>

// ---------------------------------------------------------------------------
// GAT + MLP pipeline for MI455X (gfx1250, wave32, WMMA)
//   K1: h = x @ W          -> v_wmma_f32_16x16x32_bf16, b128 LDS fragment loads
//   K2: al_src/al_dst      -> per (node,head) dot products
//   K3: segment max        -> ordered-uint atomicMax (L2-resident)
//   K4: segment sum        -> atomicAdd f32 on denom (L2-resident)
//   K5: alpha scatter-add  -> 6x atomicAdd f32 per (edge,head) into accumulator
//   K6: bias+relu, fc1+relu, lin -> one thread per node, weights in LDS
// ---------------------------------------------------------------------------

#define N_FIN   128
#define N_HEADS 8
#define N_COUT  6
#define N_HC    48   // HEADS * C_OUT

#define LDS_PAD   8
#define A_STRIDE  (N_FIN + LDS_PAD)   // 136 bf16 = 272 B  (4-bank skew / row)
#define B_STRIDE  (N_FIN + LDS_PAD)

typedef __attribute__((ext_vector_type(16))) __bf16 v16bf;
typedef __attribute__((ext_vector_type(8)))  __bf16 v8bf;   // one ds_load_b128
typedef __attribute__((ext_vector_type(4)))  __bf16 bf4;
typedef __attribute__((ext_vector_type(4)))  float  f4;
typedef __attribute__((ext_vector_type(8)))  float  v8f;

// ---- monotone ordered-uint encoding for float atomic max --------------------
__device__ __forceinline__ unsigned enc_f32(float f) {
    unsigned u = __float_as_uint(f);
    return (u & 0x80000000u) ? ~u : (u | 0x80000000u);
}
__device__ __forceinline__ float dec_f32(unsigned u) {
    return (u & 0x80000000u) ? __uint_as_float(u & 0x7fffffffu)
                             : __uint_as_float(~u);
}
#define ENC_NEG_INF 0x007FFFFFu  // enc_f32(-inf)

__device__ __forceinline__ float leaky02(float z) { return z > 0.f ? z : 0.2f * z; }

// ---------------------------------------------------------------------------
// K1: h[N,48] = x[N,128] @ W[128,48], bf16 WMMA with f32 accumulation.
// Workgroup = 4 waves, 64 rows (16 rows/wave).
// LDS: sA row-major 64x136 bf16 (17 KB), sBt = W^T 48x136 bf16 (13 KB).
// Fragment layout facts (ISA tables):
//   A  lane(m,half), elem j : K = (j<8 ? 8*half + j : 16 + 8*half + j-8)
//      -> two contiguous 8-elem runs of row m  -> 2x ds_load_b128
//   B  lane(n,half), elem j : K = (j<8 ? 8*half + j : 16 + 8*half + j-8), col fixed
//      -> two contiguous 8-elem runs along K in W^T -> 2x ds_load_b128
// ---------------------------------------------------------------------------
__global__ __launch_bounds__(128) void gemm_h_kernel(
    const float* __restrict__ x, const float* __restrict__ W,
    float* __restrict__ h, int n_nodes)
{
    __shared__ __bf16 sA[64 * A_STRIDE];
    __shared__ __bf16 sBt[N_HC * B_STRIDE];

    const int tid  = threadIdx.x;
    const int wave = tid >> 5;
    const int lane = tid & 31;
    const long row0 = (long)blockIdx.x * 64;

    // Stage W^T -> LDS (coalesced global reads, one-time scatter stores).
    for (int i = tid; i < N_FIN * N_HC; i += 128) {
        int k = i / N_HC, col = i % N_HC;
        sBt[col * B_STRIDE + k] = (__bf16)W[i];
    }
    // Stage 64 rows of x -> LDS: float4 global loads, packed bf16x4 stores.
    for (int i = tid; i < 64 * (N_FIN / 4); i += 128) {   // 16 iters exact
        int r  = i >> 5;            // 32 float4 per row
        int c4 = (i & 31) << 2;
        long gr = row0 + r;
        f4 v = {0.f, 0.f, 0.f, 0.f};
        if (gr < n_nodes) v = *reinterpret_cast<const f4*>(&x[gr * N_FIN + c4]);
        bf4 b;
        b.x = (__bf16)v.x; b.y = (__bf16)v.y;
        b.z = (__bf16)v.z; b.w = (__bf16)v.w;
        *reinterpret_cast<bf4*>(&sA[r * A_STRIDE + c4]) = b;
    }
    __syncthreads();

    const int wrow = wave << 4;
    const int m    = lane & 15;
    const int half = lane >> 4;
    const __bf16* arow = &sA[(wrow + m) * A_STRIDE];

    v8f acc[3] = {};
    #pragma unroll
    for (int s = 0; s < 4; ++s) {   // K steps of 32
        v16bf a;
        *reinterpret_cast<v8bf*>(&a) =
            *reinterpret_cast<const v8bf*>(&arow[s * 32 + 8 * half]);
        *(reinterpret_cast<v8bf*>(&a) + 1) =
            *reinterpret_cast<const v8bf*>(&arow[s * 32 + 16 + 8 * half]);
        #pragma unroll
        for (int t = 0; t < 3; ++t) {  // 3 N-tiles of 16, reuse A
            const __bf16* brow = &sBt[(t * 16 + m) * B_STRIDE + s * 32 + 8 * half];
            v16bf b;
            *reinterpret_cast<v8bf*>(&b) =
                *reinterpret_cast<const v8bf*>(&brow[0]);
            *(reinterpret_cast<v8bf*>(&b) + 1) =
                *reinterpret_cast<const v8bf*>(&brow[16]);
            acc[t] = __builtin_amdgcn_wmma_f32_16x16x32_bf16(
                false, a, false, b, (short)0, acc[t], false, false);
        }
    }

    // D layout: VGPR r -> row = r + 8*half, col = lane%16.
    #pragma unroll
    for (int t = 0; t < 3; ++t) {
        int col = t * 16 + m;
        #pragma unroll
        for (int r = 0; r < 8; ++r) {
            long grow = row0 + wrow + r + 8 * half;
            if (grow < n_nodes) h[grow * N_HC + col] = acc[t][r];
        }
    }
}

// ---------------------------------------------------------------------------
// K2: per (node,head) attention logits.
// ---------------------------------------------------------------------------
__global__ __launch_bounds__(256) void attn_logits_kernel(
    const float* __restrict__ h, const float* __restrict__ a_src,
    const float* __restrict__ a_dst,
    float* __restrict__ al_src, float* __restrict__ al_dst, int n_nodes)
{
    int idx = blockIdx.x * blockDim.x + threadIdx.x; // node*8 + head
    if (idx >= n_nodes * N_HEADS) return;
    int node = idx >> 3, head = idx & 7;
    const float* hp = h + (long)node * N_HC + head * N_COUT;
    float s = 0.f, d = 0.f;
    #pragma unroll
    for (int c = 0; c < N_COUT; ++c) {
        float v = hp[c];
        s += v * a_src[head * N_COUT + c];
        d += v * a_dst[head * N_COUT + c];
    }
    al_src[idx] = s;
    al_dst[idx] = d;
}

// ---------------------------------------------------------------------------
// K_init: m_enc = enc(-inf), denom = 0, gat accumulator = 0.
// ---------------------------------------------------------------------------
__global__ __launch_bounds__(256) void init_kernel(
    unsigned* __restrict__ m_enc, float* __restrict__ denom,
    float* __restrict__ gat, int n_nodes)
{
    long i = (long)blockIdx.x * blockDim.x + threadIdx.x;
    long nh = (long)n_nodes * N_HEADS;
    if (i < nh) { m_enc[i] = ENC_NEG_INF; denom[i] = 0.f; }
    long nc = (long)n_nodes * N_HC;
    if (i < nc) gat[i] = 0.f;
}

// ---------------------------------------------------------------------------
// K3: segment max over incoming edges (ordered-uint atomicMax).
// Thread = (edge, head). Adjacent threads share the edge -> coalesced gathers.
// ---------------------------------------------------------------------------
__global__ __launch_bounds__(256) void edge_max_kernel(
    const int* __restrict__ ei, const float* __restrict__ ew,
    const float* __restrict__ al_src, const float* __restrict__ al_dst,
    unsigned* __restrict__ m_enc, long n_edges)
{
    long idx = (long)blockIdx.x * blockDim.x + threadIdx.x;
    if (idx >= n_edges * N_HEADS) return;
    long e = idx >> 3; int head = idx & 7;
    int src = ei[e], dst = ei[n_edges + e];
    float z = leaky02(al_src[(long)src * N_HEADS + head] +
                      al_dst[(long)dst * N_HEADS + head]) * ew[e];
    atomicMax(&m_enc[(long)dst * N_HEADS + head], enc_f32(z));
}

// ---------------------------------------------------------------------------
// K4: softmax denominator (recompute logit; all gathers L2-resident).
// ---------------------------------------------------------------------------
__global__ __launch_bounds__(256) void edge_sum_kernel(
    const int* __restrict__ ei, const float* __restrict__ ew,
    const float* __restrict__ al_src, const float* __restrict__ al_dst,
    const unsigned* __restrict__ m_enc, float* __restrict__ denom, long n_edges)
{
    long idx = (long)blockIdx.x * blockDim.x + threadIdx.x;
    if (idx >= n_edges * N_HEADS) return;
    long e = idx >> 3; int head = idx & 7;
    int src = ei[e], dst = ei[n_edges + e];
    long dh = (long)dst * N_HEADS + head;
    float z = leaky02(al_src[(long)src * N_HEADS + head] + al_dst[dh]) * ew[e];
    float ex = __expf(z - dec_f32(m_enc[dh]));
    atomicAdd(&denom[dh], ex);
}

// ---------------------------------------------------------------------------
// K5: alpha-weighted message scatter-add. 6 f32 atomics per (edge,head);
// accumulator (19.2 MB) stays in L2. Prefetch edge stream ahead.
// ---------------------------------------------------------------------------
__global__ __launch_bounds__(256) void edge_scatter_kernel(
    const int* __restrict__ ei, const float* __restrict__ ew,
    const float* __restrict__ al_src, const float* __restrict__ al_dst,
    const unsigned* __restrict__ m_enc, const float* __restrict__ denom,
    const float* __restrict__ h, float* __restrict__ gat, long n_edges)
{
    long idx = (long)blockIdx.x * blockDim.x + threadIdx.x;
    if (idx >= n_edges * N_HEADS) return;
    long e = idx >> 3; int head = idx & 7;

    __builtin_prefetch(&ei[e + 8192], 0, 3);   // stream edge list ahead

    int src = ei[e], dst = ei[n_edges + e];
    long dh = (long)dst * N_HEADS + head;
    float z = leaky02(al_src[(long)src * N_HEADS + head] + al_dst[dh]) * ew[e];
    float alpha = __expf(z - dec_f32(m_enc[dh])) / fmaxf(denom[dh], 1e-16f);

    const float* hp = h   + (long)src * N_HC + head * N_COUT;
    float*       op = gat + (long)dst * N_HC + head * N_COUT;
    #pragma unroll
    for (int c = 0; c < N_COUT; ++c)
        atomicAdd(&op[c], hp[c] * alpha);
}

// ---------------------------------------------------------------------------
// K6: out = relu(relu(gat + bias) @ fc1 + fc1_b) @ lin + lin_b, per node.
// All weights staged in LDS once per block.
// ---------------------------------------------------------------------------
__global__ __launch_bounds__(256) void final_mlp_kernel(
    const float* __restrict__ gat, const float* __restrict__ bias,
    const float* __restrict__ fc1_w, const float* __restrict__ fc1_b,
    const float* __restrict__ lin_w, const float* __restrict__ lin_b,
    float* __restrict__ out, int n_nodes)
{
    __shared__ float sW1[N_HC * 10];
    __shared__ float sB1[10];
    __shared__ float sW2[10];
    __shared__ float sBias[N_HC];
    __shared__ float sb2;

    int tid = threadIdx.x;
    for (int i = tid; i < N_HC * 10; i += blockDim.x) sW1[i] = fc1_w[i];
    if (tid < 10) { sB1[tid] = fc1_b[tid]; sW2[tid] = lin_w[tid]; }
    if (tid < N_HC) sBias[tid] = bias[tid];
    if (tid == 0) sb2 = lin_b[0];
    __syncthreads();

    int node = blockIdx.x * blockDim.x + tid;
    if (node >= n_nodes) return;

    float hrow[N_HC];
    const float* gp = gat + (long)node * N_HC;
    #pragma unroll
    for (int j = 0; j < N_HC; ++j) {
        float v = gp[j] + sBias[j];
        hrow[j] = v > 0.f ? v : 0.f;
    }
    float acc2 = sb2;
    #pragma unroll
    for (int k = 0; k < 10; ++k) {
        float a = sB1[k];
        #pragma unroll
        for (int j = 0; j < N_HC; ++j) a += hrow[j] * sW1[j * 10 + k];
        a = a > 0.f ? a : 0.f;
        acc2 += a * sW2[k];
    }
    out[node] = acc2;
}

// ---------------------------------------------------------------------------
extern "C" void kernel_launch(void* const* d_in, const int* in_sizes, int n_in,
                              void* d_out, int out_size, void* d_ws, size_t ws_size,
                              hipStream_t stream)
{
    const float* x      = (const float*)d_in[0];
    const int*   ei     = (const int*)  d_in[1];   // edge_index [2,E]
    const float* ew     = (const float*)d_in[2];   // edge_weight [E]
    const float* W      = (const float*)d_in[3];
    const float* a_src  = (const float*)d_in[4];
    const float* a_dst  = (const float*)d_in[5];
    const float* bias   = (const float*)d_in[6];
    const float* fc1_w  = (const float*)d_in[7];
    const float* fc1_b  = (const float*)d_in[8];
    const float* lin_w  = (const float*)d_in[9];
    const float* lin_b  = (const float*)d_in[10];
    float*       out    = (float*)d_out;

    const int  n_nodes = in_sizes[0] / N_FIN;
    const long n_edges = (long)in_sizes[2];

    // Workspace carve-up (all f32/u32, 51.2 MB total; all L2-resident).
    float*    h      = (float*)d_ws;                       // n*48
    float*    al_src = h      + (long)n_nodes * N_HC;      // n*8
    float*    al_dst = al_src + (long)n_nodes * N_HEADS;   // n*8
    unsigned* m_enc  = (unsigned*)(al_dst + (long)n_nodes * N_HEADS); // n*8
    float*    denom  = (float*)(m_enc + (long)n_nodes * N_HEADS);     // n*8
    float*    gat    = denom  + (long)n_nodes * N_HEADS;   // n*48

    // K1: WMMA GEMM
    int gemm_blocks = (n_nodes + 63) / 64;
    gemm_h_kernel<<<gemm_blocks, 128, 0, stream>>>(x, W, h, n_nodes);

    // K2: attention logits
    int nh = n_nodes * N_HEADS;
    attn_logits_kernel<<<(nh + 255) / 256, 256, 0, stream>>>(
        h, a_src, a_dst, al_src, al_dst, n_nodes);

    // K_init
    long nc = (long)n_nodes * N_HC;
    init_kernel<<<(unsigned)((nc + 255) / 256), 256, 0, stream>>>(
        m_enc, denom, gat, n_nodes);

    // K3/K4/K5: edge passes
    long eh = n_edges * N_HEADS;
    unsigned eblocks = (unsigned)((eh + 255) / 256);
    edge_max_kernel<<<eblocks, 256, 0, stream>>>(ei, ew, al_src, al_dst, m_enc, n_edges);
    edge_sum_kernel<<<eblocks, 256, 0, stream>>>(ei, ew, al_src, al_dst, m_enc, denom, n_edges);
    edge_scatter_kernel<<<eblocks, 256, 0, stream>>>(
        ei, ew, al_src, al_dst, m_enc, denom, h, gat, n_edges);

    // K6: final MLP
    final_mlp_kernel<<<(n_nodes + 255) / 256, 256, 0, stream>>>(
        gat, bias, fc1_w, fc1_b, lin_w, lin_b, out, n_nodes);
}
